// DotProductAttention_17952963297408
// MI455X (gfx1250) — matrix operand order repaired and checked by reading, not compiled, not run on hardware
//
#include <hip/hip_runtime.h>

typedef __attribute__((ext_vector_type(16))) _Float16 v16h;
typedef __attribute__((ext_vector_type(8)))  _Float16 v8h;
typedef __attribute__((ext_vector_type(8)))  float    v8f;

#define BQ 64          // query rows per workgroup
#define BK 32          // keys per k-tile (matches WMMA K for f16)
#define NEGBIG (-1.0e30f)
#define DD 64          // head dim

// A-fragment (16-bit 16x32): half 0 -> K 0..7 / 16..23, half 1 -> K 8..15 / 24..31
__device__ __forceinline__ v16h fragA(const _Float16* row, int base) {
  v8h lo = *(const v8h*)(row + base);        // ds_load_b128
  v8h hi = *(const v8h*)(row + base + 16);   // ds_load_b128
  v16h r;
#pragma unroll
  for (int i = 0; i < 8; ++i) { r[i] = lo[i]; r[i + 8] = hi[i]; }
  return r;
}

// B-fragment (16-bit 32x16): half 0 -> K 0..7 / 16..23, half 1 -> K 8..15 / 24..31
__device__ __forceinline__ v16h fragB(const _Float16* row, int base) {
  v8h lo = *(const v8h*)(row + base);
  v8h hi = *(const v8h*)(row + base + 16);
  v16h r;
#pragma unroll
  for (int i = 0; i < 8; ++i) { r[i] = lo[i]; r[i + 8] = hi[i]; }
  return r;
}

// 16-lane max reduction: DPP feeds v_max_num_f32 directly (1 VALU per step)
__device__ __forceinline__ float redmax16(float v) {
  asm volatile("v_max_num_f32_dpp %0, %0, %0 quad_perm:[1,0,3,2] row_mask:0xf bank_mask:0xf" : "+v"(v));
  asm volatile("v_max_num_f32_dpp %0, %0, %0 quad_perm:[2,3,0,1] row_mask:0xf bank_mask:0xf" : "+v"(v));
  asm volatile("v_max_num_f32_dpp %0, %0, %0 row_half_mirror row_mask:0xf bank_mask:0xf" : "+v"(v));
  asm volatile("v_max_num_f32_dpp %0, %0, %0 row_mirror row_mask:0xf bank_mask:0xf" : "+v"(v));
  return v;
}

// raw v_exp_f32: no libm denormal fixup (args <= 0; deep negatives flush to 0 as desired)
__device__ __forceinline__ float fast_exp2(float x) { return __builtin_amdgcn_exp2f(x); }

// next K/V tile -> registers (issued early to hide HBM latency)
__device__ __forceinline__ void tile_load(const float* kgb, const float* vgb,
                                          int kt, int kend, int tid,
                                          float4 f[4], float4 g[4]) {
#pragma unroll
  for (int i = 0; i < 4; ++i) {
    int idx = tid + i * 128;            // 0..511 float4s
    int kk  = idx >> 4;
    int d4  = (idx & 15) << 2;
    float4 z = make_float4(0.f, 0.f, 0.f, 0.f);
    f[i] = z; g[i] = z;
    if (kt + kk < kend) {
      f[i] = *(const float4*)(kgb + (size_t)(kt + kk) * DD + d4);
      g[i] = *(const float4*)(vgb + (size_t)(kt + kk) * DD + d4);
    }
  }
}

// registers -> f16 LDS (K row-major, V transposed)
__device__ __forceinline__ void tile_store(_Float16 (*Kb)[DD], _Float16 (*Vb)[BK],
                                           int tid, const float4 f[4], const float4 g[4]) {
#pragma unroll
  for (int i = 0; i < 4; ++i) {
    int idx = tid + i * 128;
    int kk  = idx >> 4;
    int d4  = (idx & 15) << 2;
    Kb[kk][d4 + 0] = (_Float16)f[i].x;
    Kb[kk][d4 + 1] = (_Float16)f[i].y;
    Kb[kk][d4 + 2] = (_Float16)f[i].z;
    Kb[kk][d4 + 3] = (_Float16)f[i].w;
    Vb[d4 + 0][kk] = (_Float16)g[i].x;
    Vb[d4 + 1][kk] = (_Float16)g[i].y;
    Vb[d4 + 2][kk] = (_Float16)g[i].z;
    Vb[d4 + 3][kk] = (_Float16)g[i].w;
  }
}

__global__ __launch_bounds__(128)
void fa_causal_vlen_kernel(const float* __restrict__ Q, const float* __restrict__ K,
                           const float* __restrict__ V, const int* __restrict__ VL,
                           float* __restrict__ Out, int Hc, int Lc) {
  const int ntile = Lc / BQ;
  const int qt = blockIdx.x % ntile;
  const int bh = blockIdx.x / ntile;
  const int b  = bh / Hc;
  const int h  = bh % Hc;
  const int q0 = qt * BQ;

  __shared__ _Float16 Qh[BQ][DD];          // 8 KB
  __shared__ _Float16 Kh[2][BK][DD];       // 8 KB  double-buffered [key][d]
  __shared__ _Float16 Vt[2][DD][BK];       // 8 KB  double-buffered [d][key]
  __shared__ _Float16 Ph[4][16][BK];       // 4 KB  per-wave P staging

  const int tid   = threadIdx.x;
  const int wave  = tid >> 5;
  const int lane  = tid & 31;
  const int lhalf = lane >> 4;
  const int lcol  = lane & 15;
  const int abase = lhalf * 8;    // A-fragment per-half base
  const int bbase = lhalf * 8;    // B-fragment per-half base

  // softmax runs in exp2 domain: fold 1/sqrt(D) * log2(e) into Q once
  const float qscale = 0.125f * 1.44269504088896f;

  int vlen = VL[b];
  vlen = vlen < 1 ? 1 : (vlen > Lc ? Lc : vlen);
  const int qend = q0 + BQ;
  const int kend = qend < vlen ? qend : vlen;   // union of columns any row needs
  const int nt   = (kend + BK - 1) / BK;

  const float* qg  = Q + ((size_t)bh * Lc + q0) * DD;
  const float* kgb = K + (size_t)bh * Lc * DD;
  const float* vgb = V + (size_t)bh * Lc * DD;

  // ---- Q tile -> f16 LDS, pre-scaled into exp2 domain ----
#pragma unroll
  for (int i = 0; i < 8; ++i) {
    int idx = tid + i * 128;
    int r  = idx >> 4;
    int d4 = (idx & 15) << 2;
    float4 f = *(const float4*)(qg + r * DD + d4);
    Qh[r][d4 + 0] = (_Float16)(f.x * qscale);
    Qh[r][d4 + 1] = (_Float16)(f.y * qscale);
    Qh[r][d4 + 2] = (_Float16)(f.z * qscale);
    Qh[r][d4 + 3] = (_Float16)(f.w * qscale);
  }

  // ---- prologue: tile 0 -> regs -> LDS buffer 0 ----
  float4 fr[4], gr[4];
  tile_load(kgb, vgb, 0, kend, tid, fr, gr);
  tile_store(Kh[0], Vt[0], tid, fr, gr);
  __syncthreads();

  // Q A-fragments for this wave (row M = lcol), two 32-deep K chunks of D
  const _Float16* qrow = &Qh[wave * 16 + lcol][0];
  v16h qa0 = fragA(qrow, 0 + abase);
  v16h qa1 = fragA(qrow, 32 + abase);

  // constant all-ones 32x16 B-matrix: rowsum(P) = P * ones via one WMMA
  v16h onesB;
#pragma unroll
  for (int i = 0; i < 16; ++i) onesB[i] = (_Float16)1.0f;

  v8f acc[4];
#pragma unroll
  for (int c = 0; c < 4; ++c) { v8f z = {}; acc[c] = z; }
  float mrow[8], lrow[8];
#pragma unroll
  for (int r = 0; r < 8; ++r) { mrow[r] = NEGBIG; lrow[r] = 0.0f; }

  const int row0w = q0 + wave * 16;             // min row of this wave's tile
  const int row0  = row0w + lhalf * 8;          // global row of acc element r

  for (int t = 0; t < nt; ++t) {
    const int kt  = t * BK;
    const int cur = t & 1;
    const bool more = (t + 1 < nt);

    if (t + 2 < nt) {   // warm L2 two tiles ahead -> global_prefetch_b8
      __builtin_prefetch(kgb + (size_t)(kt + 2 * BK) * DD + tid * 16, 0, 1);
      __builtin_prefetch(vgb + (size_t)(kt + 2 * BK) * DD + tid * 16, 0, 1);
    }
    if (more) tile_load(kgb, vgb, kt + BK, kend, tid, fr, gr);  // overlap with compute

    // ---- S = Q * K^T for two 16-key subtiles (K-dim = D = 2 x 32) ----
    v8f s[2];
#pragma unroll
    for (int ns = 0; ns < 2; ++ns) {
      const _Float16* krow = &Kh[cur][ns * 16 + lcol][0];
      v16h kb0 = fragB(krow, 0 + bbase);
      v16h kb1 = fragB(krow, 32 + bbase);
      v8f z = {};
      z = __builtin_amdgcn_wmma_f32_16x16x32_f16(false, qa0, false, kb0, (short)0, z, false, false);
      z = __builtin_amdgcn_wmma_f32_16x16x32_f16(false, qa1, false, kb1, (short)0, z, false, false);
      s[ns] = z;
    }

    // wave-uniform fast path: tile fully below diagonal and inside valid_len
    const int nomask = __builtin_amdgcn_readfirstlane(
        (kt + BK <= row0w + 1 && kt + BK <= vlen) ? 1 : 0);

    // ---- online softmax in exp2 domain (row-max via DPP-fused v_max) ----
    float corr[8];
#pragma unroll
    for (int r = 0; r < 8; ++r) {
      float s0 = s[0][r];
      float s1 = s[1][r];
      if (!nomask) {
        const int row = row0 + r;
        const int c0 = kt + lcol;
        const int c1 = c0 + 16;
        const bool rowinv = (row >= vlen);
        if (!((c0 < vlen) && (rowinv || c0 <= row))) s0 = NEGBIG;
        if (!((c1 < vlen) && (rowinv || c1 <= row))) s1 = NEGBIG;
      }
      float tmax = redmax16(fmaxf(s0, s1));
      float mnew = fmaxf(mrow[r], tmax);
      float cf = fast_exp2(mrow[r] - mnew);
      float p0 = fast_exp2(s0 - mnew);
      float p1 = fast_exp2(s1 - mnew);
      mrow[r] = mnew;
      corr[r] = cf;
      Ph[wave][r + lhalf * 8][lcol]      = (_Float16)p0;
      Ph[wave][r + lhalf * 8][lcol + 16] = (_Float16)p1;
    }
#pragma unroll
    for (int c = 0; c < 4; ++c)
#pragma unroll
      for (int r = 0; r < 8; ++r) acc[c][r] *= corr[r];

    // same-wave LDS RAW on Ph: DS ops are in-order; fence counter + compiler
    asm volatile("s_wait_dscnt 0x0" ::: "memory");

    // ---- lsum = P * ones (row sums, broadcast across lanes); O += P * V ----
    v16h pa = fragA(&Ph[wave][lcol][0], abase);
    v8f zz = {};
    v8f lsum = __builtin_amdgcn_wmma_f32_16x16x32_f16(false, pa, false, onesB, (short)0, zz, false, false);
#pragma unroll
    for (int c = 0; c < 4; ++c) {
      v16h vb = fragB(&Vt[cur][c * 16 + lcol][0], bbase);
      acc[c] = __builtin_amdgcn_wmma_f32_16x16x32_f16(false, pa, false, vb, (short)0, acc[c], false, false);
    }
#pragma unroll
    for (int r = 0; r < 8; ++r) lrow[r] = lrow[r] * corr[r] + lsum[r];

    if (more) {
      tile_store(Kh[cur ^ 1], Vt[cur ^ 1], tid, fr, gr);
      __syncthreads();   // one barrier per iteration (double-buffered)
    }
  }

  // ---- epilogue: normalize and store [B, L, H*D] fp32 ----
#pragma unroll
  for (int r = 0; r < 8; ++r) {
    const float inv = 1.0f / lrow[r];
    const int row = row0 + r;
    float* op = Out + ((size_t)b * Lc + row) * (size_t)(Hc * DD) + h * DD + lcol;
#pragma unroll
    for (int c = 0; c < 4; ++c) op[c * 16] = acc[c][r] * inv;
  }
}

extern "C" void kernel_launch(void* const* d_in, const int* in_sizes, int n_in,
                              void* d_out, int out_size, void* d_ws, size_t ws_size,
                              hipStream_t stream) {
  (void)in_sizes; (void)n_in; (void)out_size; (void)d_ws; (void)ws_size;
  const float* q = (const float*)d_in[0];
  const float* k = (const float*)d_in[1];
  const float* v = (const float*)d_in[2];
  // d_in[3] is the 2048x2048 additive causal mask: applied analytically, never read
  const int* vl  = (const int*)d_in[4];
  float* out = (float*)d_out;

  const int B = 2, H = 16, L = 2048;
  dim3 grid(B * H * (L / BQ));
  dim3 block(128);
  hipLaunchKernelGGL(fa_causal_vlen_kernel, grid, block, 0, stream, q, k, v, vl, out, H, L);
}
